// HierarchicalSpiralMamba_19207093747905
// MI455X (gfx1250) — matrix operand order repaired and checked by reading, not compiled, not run on hardware
//
#include <hip/hip_runtime.h>
#include <hip/hip_bf16.h>

typedef __attribute__((ext_vector_type(16))) _Float16 v16h;
typedef __attribute__((ext_vector_type(8)))  _Float16 h8;
typedef __attribute__((ext_vector_type(8)))  float    v8f;

// ---------------------------------------------------------------------------
// WMMA GEMM with N-register-blocking:
//   D[M,N] (f32) = A[M,K](f16) * W[N,K]^T(f16) (+bias, +act)
// One wave computes a 16 x (16*NB) strip: the A fragment is loaded once per
// 32-wide K step and reused across NB B fragments -> NB v_wmma issues per
// 2+2*NB b128 loads (vs 4 loads per wmma unblocked).
// Fragment layouts (v_wmma_f32_16x16x32_f16, wave32):
//   A (16x32): lane ln holds row tm+(ln&15); halves j = K k0+(ln>>4)*16+j.
//   B (32x16): lane ln holds col tn+(ln&15); halves j = K k0+(ln>>4)*16+j.
// act: 0 = none, 1 = softplus (dt_proj epilogue)
// ---------------------------------------------------------------------------
template <int NB>
__global__ void wmma_gemm_kernel(const _Float16* __restrict__ A,
                                 const _Float16* __restrict__ W,
                                 float* __restrict__ D,
                                 const float* __restrict__ bias,
                                 int M, int N, int K, int act) {
  const int lane = threadIdx.x & 31;
  const int wib  = threadIdx.x >> 5;
  const int groups_n = N / (16 * NB);
  const int tile = blockIdx.x * (blockDim.x >> 5) + wib;
  const int total = (M >> 4) * groups_n;
  if (tile >= total) return;
  const int tm  = (tile / groups_n) << 4;
  const int tn0 = (tile % groups_n) * (16 * NB);
  const int hi  = lane >> 4;
  const int row = tm + (lane & 15);

  v8f acc[NB];
#pragma unroll
  for (int nb = 0; nb < NB; ++nb) acc[nb] = v8f{};

  const _Float16* aRow = A + (size_t)row * K + (hi << 4);
  const _Float16* wRow = W + (size_t)(tn0 + (lane & 15)) * K + (hi << 4);
  const size_t wStride = (size_t)16 * K;   // halves between adjacent N-tiles

  for (int k0 = 0; k0 < K; k0 += 32) {
    h8 a0 = *(const h8*)(aRow + k0);
    h8 a1 = *(const h8*)(aRow + k0 + 8);
    v16h af;
#pragma unroll
    for (int i = 0; i < 8; ++i) { af[i] = a0[i]; af[8 + i] = a1[i]; }
#pragma unroll
    for (int nb = 0; nb < NB; ++nb) {
      const _Float16* wp = wRow + nb * wStride + k0;
      h8 b0 = *(const h8*)(wp);
      h8 b1 = *(const h8*)(wp + 8);
      v16h bf;
#pragma unroll
      for (int i = 0; i < 8; ++i) { bf[i] = b0[i]; bf[8 + i] = b1[i]; }
      acc[nb] = __builtin_amdgcn_wmma_f32_16x16x32_f16(
          false, af, false, bf, (short)0, acc[nb], false, false);
    }
  }

#pragma unroll
  for (int nb = 0; nb < NB; ++nb) {
    const int col = tn0 + nb * 16 + (lane & 15);
    const float bv = bias ? bias[col] : 0.0f;
#pragma unroll
    for (int v = 0; v < 8; ++v) {
      int m = tm + v + (hi << 3);
      float r = acc[nb][v] + bv;
      if (act == 1) r = (r > 20.f) ? r : __logf(1.f + __expf(r));
      D[(size_t)m * N + col] = r;
    }
  }
}

// ---- spiral gather + layernorm -> f16 sequence (b, i, c) --------------------
__global__ void ln_gather_kernel(const float* __restrict__ x,   // (B,256,L)
                                 const int* __restrict__ perm,
                                 const float* __restrict__ lw,
                                 const float* __restrict__ lb,
                                 _Float16* __restrict__ seq, int L, int rev) {
  int i = blockIdx.x % L, bb = blockIdx.x / L;
  int c = threadIdx.x;                      // 256 threads
  int si = rev ? (L - 1 - i) : i;
  int pos = perm[si];
  float v = x[((size_t)bb * 256 + c) * L + pos];
  __shared__ float red[256];
  red[c] = v; __syncthreads();
  for (int s = 128; s > 0; s >>= 1) { if (c < s) red[c] += red[c + s]; __syncthreads(); }
  float mean = red[0] / 256.f; __syncthreads();
  float d = v - mean;
  red[c] = d * d; __syncthreads();
  for (int s = 128; s > 0; s >>= 1) { if (c < s) red[c] += red[c + s]; __syncthreads(); }
  float var = red[0] / 256.f;
  float y = d * rsqrtf(var + 1e-5f) * lw[c] + lb[c];
  seq[((size_t)(bb * L + i)) * 256 + c] = (_Float16)y;
}

// ---- depthwise causal conv (D_CONV=4) + SiLU -> f16 (b,i,d) -----------------
__global__ void conv_silu_kernel(const float* __restrict__ xz,  // (b,i,1024), xi = [:512]
                                 const float* __restrict__ cw,  // (512,4)
                                 const float* __restrict__ cb,
                                 _Float16* __restrict__ xc16, int L) {
  size_t idx = (size_t)blockIdx.x * 256 + threadIdx.x;
  size_t tot = (size_t)2 * L * 512;
  if (idx >= tot) return;
  int d = idx % 512; size_t m = idx / 512;
  int i = (int)(m % L), bb = (int)(m / L);
  float acc = cb[d];
#pragma unroll
  for (int k = 0; k < 4; ++k) {
    int j = i - 3 + k;
    if (j >= 0) acc += cw[d * 4 + k] * xz[((size_t)(bb * L + j)) * 1024 + d];
  }
  float s = acc / (1.f + __expf(-acc));
  xc16[m * 512 + d] = (_Float16)s;
}

// ---- pack dt columns of dbl (L,48) into K-padded f16 (L,32) -----------------
__global__ void dtin_pack_kernel(const float* __restrict__ dbl,
                                 _Float16* __restrict__ dtin, size_t n) {
  size_t idx = (size_t)blockIdx.x * 256 + threadIdx.x;
  if (idx >= n) return;
  int j = idx & 31; size_t m = idx >> 5;
  dtin[idx] = (_Float16)((j < 16) ? dbl[m * 48 + j] : 0.f);
}

// ---- selective scan: 512 threads, one channel each, 16 states in regs -------
__global__ void scan_kernel(const float* __restrict__ dt,     // (b,i,512)
                            const float* __restrict__ dbl,    // (b,i,48): B=16..31,C=32..47
                            const _Float16* __restrict__ xc16,// (b,i,512)
                            const float* __restrict__ xz,     // (b,i,1024): z = [512:]
                            const float* __restrict__ A_log,  // (512,16)
                            const float* __restrict__ Dp,     // (512,)
                            _Float16* __restrict__ ypack, int L) {
  int bb = blockIdx.x;
  int d = threadIdx.x;                       // 512 threads
  float A[16], h[16];
#pragma unroll
  for (int s = 0; s < 16; ++s) { A[s] = -__expf(A_log[d * 16 + s]); h[s] = 0.f; }
  __shared__ float sB[16], sC[16];
  size_t base = (size_t)bb * L;
  float Dv = Dp[d];
  for (int i = 0; i < L; ++i) {
    size_t m = base + i;
    if (d < 32) {
      float v = dbl[m * 48 + 16 + d];
      if (d < 16) sB[d] = v; else sC[d - 16] = v;
    }
    __syncthreads();
    float dtv = dt[m * 512 + d];
    float xv  = (float)xc16[m * 512 + d];
    float dx  = dtv * xv;
    float y = 0.f;
#pragma unroll
    for (int s = 0; s < 16; ++s) {
      float dA = __expf(dtv * A[s]);
      h[s] = dA * h[s] + dx * sB[s];
      y += h[s] * sC[s];
    }
    float z = xz[m * 1024 + 512 + d];
    float g = z / (1.f + __expf(-z));
    ypack[m * 512 + d] = (_Float16)((y + xv * Dv) * g);
    __syncthreads();
  }
}

// ---- misc pack / permute ----------------------------------------------------
__global__ void inv_perm_kernel(const int* __restrict__ perm, int* __restrict__ inv, int L) {
  int i = blockIdx.x * 256 + threadIdx.x;
  if (i < L) inv[perm[i]] = i;
}

__global__ void fuse_pack_kernel(const float* __restrict__ o1, const float* __restrict__ o2,
                                 const int* __restrict__ inv, _Float16* __restrict__ Af, int L) {
  size_t idx = (size_t)blockIdx.x * 256 + threadIdx.x;
  size_t tot = (size_t)2 * L * 512;
  if (idx >= tot) return;
  int c = idx % 512; size_t m = idx / 512;
  int p = (int)(m % L), bb = (int)(m / L);
  int i1 = inv[p];
  float v = (c < 256) ? o1[((size_t)bb * L + i1) * 256 + c]
                      : o2[((size_t)bb * L + (L - 1 - i1)) * 256 + (c - 256)];
  Af[idx] = (_Float16)v;
}

__global__ void cat_pack_kernel(const float* __restrict__ a, const float* __restrict__ b,
                                _Float16* __restrict__ dst, size_t rows) {
  size_t idx = (size_t)blockIdx.x * 256 + threadIdx.x;
  size_t tot = rows * 512;
  if (idx >= tot) return;
  int c = idx % 512; size_t m = idx / 512;
  dst[idx] = (_Float16)((c < 256) ? a[m * 256 + c] : b[m * 256 + c - 256]);
}

// ---- bilinear resize, (b, p, c=256) layout ----------------------------------
__global__ void resize_kernel(const float* __restrict__ src, float* __restrict__ dst,
                              int H1, int W1, int H2, int W2) {
  size_t idx = (size_t)blockIdx.x * 256 + threadIdx.x;
  size_t tot = (size_t)2 * H2 * W2 * 256;
  if (idx >= tot) return;
  int c = idx % 256; size_t m = idx / 256;
  int p = (int)(m % (H2 * W2)), bb = (int)(m / (H2 * W2));
  int y2 = p / W2, x2 = p % W2;
  float fy = (y2 + 0.5f) * (float)H1 / (float)H2 - 0.5f;
  float fx = (x2 + 0.5f) * (float)W1 / (float)W2 - 0.5f;
  fy = fminf(fmaxf(fy, 0.f), (float)(H1 - 1));
  fx = fminf(fmaxf(fx, 0.f), (float)(W1 - 1));
  int y0 = (int)fy, x0 = (int)fx;
  int y1 = min(y0 + 1, H1 - 1), x1 = min(x0 + 1, W1 - 1);
  float ay = fy - y0, ax = fx - x0;
  const float* sb = src + (size_t)bb * H1 * W1 * 256 + c;
  float v00 = sb[(size_t)(y0 * W1 + x0) * 256];
  float v01 = sb[(size_t)(y0 * W1 + x1) * 256];
  float v10 = sb[(size_t)(y1 * W1 + x0) * 256];
  float v11 = sb[(size_t)(y1 * W1 + x1) * 256];
  dst[idx] = (1 - ay) * ((1 - ax) * v00 + ax * v01) + ay * ((1 - ax) * v10 + ax * v11);
}

// ---- batchnorm (inference) + relu ------------------------------------------
__global__ void bn_relu_kernel(const float* __restrict__ t,
                               const float* __restrict__ mean, const float* __restrict__ var,
                               const float* __restrict__ gamma, const float* __restrict__ beta,
                               float* __restrict__ out, size_t n) {
  size_t idx = (size_t)blockIdx.x * 256 + threadIdx.x;
  if (idx >= n) return;
  int c = idx % 256;
  float v = (t[idx] - mean[c]) * rsqrtf(var[c] + 1e-5f);
  v = v * gamma[c] + beta[c];
  out[idx] = v > 0.f ? v : 0.f;
}

// ---- bilinear resample to 32x32 + spatial mean -> pooled slice --------------
__global__ void pool_kernel(const float* __restrict__ enh, float* __restrict__ pooled,
                            int H1, int W1, int lvl) {
  int c = blockIdx.x % 256, bb = blockIdx.x / 256;
  int t = threadIdx.x;                       // 256 threads, 4 pixels each
  const float* sb = enh + (size_t)bb * H1 * W1 * 256 + c;
  float sum = 0.f;
  for (int p = t; p < 1024; p += 256) {
    int y2 = p >> 5, x2 = p & 31;
    float fy = (y2 + 0.5f) * (float)H1 / 32.f - 0.5f;
    float fx = (x2 + 0.5f) * (float)W1 / 32.f - 0.5f;
    fy = fminf(fmaxf(fy, 0.f), (float)(H1 - 1));
    fx = fminf(fmaxf(fx, 0.f), (float)(W1 - 1));
    int y0 = (int)fy, x0 = (int)fx;
    int y1 = min(y0 + 1, H1 - 1), x1 = min(x0 + 1, W1 - 1);
    float ay = fy - y0, ax = fx - x0;
    float v00 = sb[(size_t)(y0 * W1 + x0) * 256];
    float v01 = sb[(size_t)(y0 * W1 + x1) * 256];
    float v10 = sb[(size_t)(y1 * W1 + x0) * 256];
    float v11 = sb[(size_t)(y1 * W1 + x1) * 256];
    sum += (1 - ay) * ((1 - ax) * v00 + ax * v01) + ay * ((1 - ax) * v10 + ax * v11);
  }
  __shared__ float red[256];
  red[t] = sum; __syncthreads();
  for (int s = 128; s > 0; s >>= 1) { if (t < s) red[t] += red[t + s]; __syncthreads(); }
  if (t == 0) pooled[bb * 1024 + lvl * 256 + c] = red[0] / 1024.f;
}

// ---- global attention MLP: a = sigmoid(relu(p@W1^T+b1)@W2^T+b2) -------------
__global__ void ga_kernel(const float* __restrict__ pooled,
                          const float* __restrict__ w1, const float* __restrict__ b1,
                          const float* __restrict__ w2, const float* __restrict__ b2,
                          float* __restrict__ att) {
  int t = threadIdx.x;                        // 256 threads
  __shared__ float hid[256];
  for (int bb = 0; bb < 2; ++bb) {
    float acc = b1[t];
    for (int j = 0; j < 1024; ++j) acc += pooled[bb * 1024 + j] * w1[t * 1024 + j];
    hid[t] = acc > 0.f ? acc : 0.f;
    __syncthreads();
    float a2 = b2[t];
    for (int j = 0; j < 256; ++j) a2 += hid[j] * w2[t * 256 + j];
    att[bb * 256 + t] = 1.f / (1.f + __expf(-a2));
    __syncthreads();
  }
}

// ---- final: (b,p,c) -> (b,c,p) channel-major, scaled by attention -----------
__global__ void out_scale_kernel(const float* __restrict__ enh, const float* __restrict__ att,
                                 float* __restrict__ out, int P) {
  size_t idx = (size_t)blockIdx.x * 256 + threadIdx.x;
  size_t tot = (size_t)2 * 256 * P;
  if (idx >= tot) return;
  int p = idx % P; size_t r = idx / P;
  int c = (int)(r % 256), bb = (int)(r / 256);
  out[idx] = enh[((size_t)bb * P + p) * 256 + c] * att[bb * 256 + c];
}

// ---- weight conversion ------------------------------------------------------
__global__ void cvt_f16_kernel(const float* __restrict__ s, _Float16* __restrict__ d, size_t n) {
  size_t i = (size_t)blockIdx.x * 256 + threadIdx.x;
  if (i < n) d[i] = (_Float16)s[i];
}
__global__ void dtw_pad_kernel(const float* __restrict__ w, _Float16* __restrict__ d) {
  int idx = blockIdx.x * 256 + threadIdx.x;     // 512*32
  if (idx >= 512 * 32) return;
  int k = idx & 31, ch = idx >> 5;
  d[idx] = (_Float16)((k < 16) ? w[ch * 16 + k] : 0.f);
}

// ===========================================================================
extern "C" void kernel_launch(void* const* d_in, const int* in_sizes, int n_in,
                              void* d_out, int out_size, void* d_ws, size_t ws_size,
                              hipStream_t stream) {
  (void)in_sizes; (void)n_in; (void)out_size; (void)ws_size;
  const int SZ[4] = {32, 16, 8, 4};
  // pytree-flatten input indexing (dict keys sorted)
  auto lvlBase   = [](int l) { return 23 + 22 * l; };    // fuse_b,fuse_w,ln_b,ln_w,m1(9),m2(9)
  auto crossBase = [](int i) { return 4 + 5 * i; };      // beta,gamma,mean,var,w
  const int GA_B1 = 19, GA_B2 = 20, GA_W1 = 21, GA_W2 = 22, PERM0 = 111;

  char* wp = (char*)d_ws;
  auto alloc = [&](size_t bytes) -> void* {
    void* p = (void*)wp; wp += (bytes + 255) & ~(size_t)255; return p;
  };
  const size_t BLm = 2 * 1024;

  _Float16 *w_in[4][2], *w_out[4][2], *w_x[4][2], *w_dt[4][2], *w_fuse[4], *w_cross[3];
  for (int l = 0; l < 4; ++l) {
    for (int dd = 0; dd < 2; ++dd) {
      w_in[l][dd]  = (_Float16*)alloc((size_t)1024 * 256 * 2);
      w_out[l][dd] = (_Float16*)alloc((size_t)256 * 512 * 2);
      w_x[l][dd]   = (_Float16*)alloc((size_t)48 * 512 * 2);
      w_dt[l][dd]  = (_Float16*)alloc((size_t)512 * 32 * 2);
    }
    w_fuse[l] = (_Float16*)alloc((size_t)256 * 512 * 2);
  }
  for (int i = 0; i < 3; ++i) w_cross[i] = (_Float16*)alloc((size_t)256 * 512 * 2);

  _Float16* seq16   = (_Float16*)alloc(BLm * 256 * 2);
  float*    xz      = (float*)   alloc(BLm * 1024 * 4);
  _Float16* xc16    = (_Float16*)alloc(BLm * 512 * 2);
  float*    dbl     = (float*)   alloc(BLm * 48 * 4);
  _Float16* dtin16  = (_Float16*)alloc(BLm * 32 * 2);
  float*    dtb     = (float*)   alloc(BLm * 512 * 4);
  _Float16* ypack16 = (_Float16*)alloc(BLm * 512 * 2);
  float*    obuf[2] = {(float*)alloc(BLm * 256 * 4), (float*)alloc(BLm * 256 * 4)};
  _Float16* Afuse   = (_Float16*)alloc(BLm * 512 * 2);
  float* mf[4], *enh[4];
  for (int l = 0; l < 4; ++l) mf[l] = (float*)alloc((size_t)2 * SZ[l] * SZ[l] * 256 * 4);
  enh[0] = mf[0];
  for (int l = 1; l < 4; ++l) enh[l] = (float*)alloc((size_t)2 * SZ[l] * SZ[l] * 256 * 4);
  float*    prev   = (float*)   alloc(BLm * 256 * 4);
  _Float16* Across = (_Float16*)alloc(BLm * 512 * 2);
  float*    crossT = (float*)   alloc(BLm * 256 * 4);
  int*      invb   = (int*)     alloc(1024 * 4);
  float*    pooled = (float*)   alloc(2 * 1024 * 4);
  float*    att    = (float*)   alloc(2 * 256 * 4);

  auto CVT = [&](const void* s, _Float16* d, size_t n) {
    cvt_f16_kernel<<<(unsigned)((n + 255) / 256), 256, 0, stream>>>((const float*)s, d, n);
  };
  auto GEMM = [&](const _Float16* A, const _Float16* W, float* D, const float* bias,
                  int M, int N, int K, int act) {
    if (N % 64 == 0) {
      int total = (M / 16) * (N / 64);
      wmma_gemm_kernel<4><<<(total + 3) / 4, 128, 0, stream>>>(A, W, D, bias, M, N, K, act);
    } else if (N % 48 == 0) {
      int total = (M / 16) * (N / 48);
      wmma_gemm_kernel<3><<<(total + 3) / 4, 128, 0, stream>>>(A, W, D, bias, M, N, K, act);
    } else {
      int total = (M / 16) * (N / 16);
      wmma_gemm_kernel<1><<<(total + 3) / 4, 128, 0, stream>>>(A, W, D, bias, M, N, K, act);
    }
  };

  // --- f16 weight staging ---------------------------------------------------
  for (int l = 0; l < 4; ++l) {
    int lb = lvlBase(l);
    CVT(d_in[lb + 1], w_fuse[l], (size_t)256 * 512);
    for (int dd = 0; dd < 2; ++dd) {
      int mb = lb + 4 + 9 * dd;
      CVT(d_in[mb + 6], w_in[l][dd],  (size_t)1024 * 256);
      CVT(d_in[mb + 7], w_out[l][dd], (size_t)256 * 512);
      CVT(d_in[mb + 8], w_x[l][dd],   (size_t)48 * 512);
      dtw_pad_kernel<<<64, 256, 0, stream>>>((const float*)d_in[mb + 5], w_dt[l][dd]);
    }
  }
  for (int i = 0; i < 3; ++i) CVT(d_in[crossBase(i) + 4], w_cross[i], (size_t)256 * 512);

  // --- per-level pipeline ---------------------------------------------------
  for (int l = 0; l < 4; ++l) {
    int S = SZ[l], L = S * S, BL = 2 * L;
    int lb = lvlBase(l);
    const float* feat = (const float*)d_in[l];
    const int*   perm = (const int*)d_in[PERM0 + l];
    const float* ln_b = (const float*)d_in[lb + 2];
    const float* ln_w = (const float*)d_in[lb + 3];

    for (int dd = 0; dd < 2; ++dd) {
      int mb = lb + 4 + 9 * dd;
      ln_gather_kernel<<<BL, 256, 0, stream>>>(feat, perm, ln_w, ln_b, seq16, L, dd);
      GEMM(seq16, w_in[l][dd], xz, nullptr, BL, 1024, 256, 0);
      conv_silu_kernel<<<(unsigned)(((size_t)BL * 512 + 255) / 256), 256, 0, stream>>>(
          xz, (const float*)d_in[mb + 3], (const float*)d_in[mb + 2], xc16, L);
      GEMM(xc16, w_x[l][dd], dbl, nullptr, BL, 48, 512, 0);
      dtin_pack_kernel<<<(unsigned)(((size_t)BL * 32 + 255) / 256), 256, 0, stream>>>(
          dbl, dtin16, (size_t)BL * 32);
      GEMM(dtin16, w_dt[l][dd], dtb, (const float*)d_in[mb + 4], BL, 512, 32, 1);
      scan_kernel<<<2, 512, 0, stream>>>(dtb, dbl, xc16, xz,
          (const float*)d_in[mb + 0], (const float*)d_in[mb + 1], ypack16, L);
      GEMM(ypack16, w_out[l][dd], obuf[dd], nullptr, BL, 256, 512, 0);
    }
    inv_perm_kernel<<<(L + 255) / 256, 256, 0, stream>>>(perm, invb, L);
    fuse_pack_kernel<<<(unsigned)(((size_t)BL * 512 + 255) / 256), 256, 0, stream>>>(
        obuf[0], obuf[1], invb, Afuse, L);
    GEMM(Afuse, w_fuse[l], mf[l], (const float*)d_in[lb + 0], BL, 256, 512, 0);

    if (l > 0) {
      int Sp = SZ[l - 1];
      resize_kernel<<<(unsigned)(((size_t)BL * 256 + 255) / 256), 256, 0, stream>>>(
          mf[l - 1], prev, Sp, Sp, S, S);
      cat_pack_kernel<<<(unsigned)(((size_t)BL * 512 + 255) / 256), 256, 0, stream>>>(
          mf[l], prev, Across, (size_t)BL);
      GEMM(Across, w_cross[l - 1], crossT, nullptr, BL, 256, 512, 0);
      int cb = crossBase(l - 1);
      bn_relu_kernel<<<(unsigned)(((size_t)BL * 256 + 255) / 256), 256, 0, stream>>>(
          crossT, (const float*)d_in[cb + 2], (const float*)d_in[cb + 3],
          (const float*)d_in[cb + 1], (const float*)d_in[cb + 0], enh[l], (size_t)BL * 256);
    }
  }

  // --- global attention + output --------------------------------------------
  for (int l = 0; l < 4; ++l)
    pool_kernel<<<512, 256, 0, stream>>>(enh[l], pooled, SZ[l], SZ[l], l);
  ga_kernel<<<1, 256, 0, stream>>>(pooled, (const float*)d_in[GA_W1], (const float*)d_in[GA_B1],
                                   (const float*)d_in[GA_W2], (const float*)d_in[GA_B2], att);
  size_t off = 0;
  for (int l = 0; l < 4; ++l) {
    int P = SZ[l] * SZ[l];
    out_scale_kernel<<<(unsigned)(((size_t)2 * 256 * P + 255) / 256), 256, 0, stream>>>(
        enh[l], att, (float*)d_out + off, P);
    off += (size_t)2 * 256 * P;
  }
}
